// Qwen3Experts_32495722561888
// MI455X (gfx1250) — compile-verified, matrix operand order checked by reading, not run on hardware
//
#include <hip/hip_runtime.h>

#define TOKENS   4096
#define NEXP     64
#define TOPK     8
#define HID      2048
#define INTER    768
#define ROWS     (TOKENS * TOPK)          // 32768 grouped rows
#define ROWS_PAD (ROWS + NEXP * 32)       // 34816 (tile-padded)
#define MAX_TILES (ROWS_PAD / 32)         // 1088 worst-case M-tiles
#define NTHREADS 256

typedef __attribute__((ext_vector_type(8)))  __bf16 v8bf;
typedef __attribute__((ext_vector_type(16))) __bf16 v16bf;
typedef __attribute__((ext_vector_type(8)))  float  v8f;

// fp32 -> bf16 round-to-nearest-even (bit-level, stored as ushort)
__device__ __forceinline__ unsigned short f2bf(float f) {
    unsigned u = __float_as_uint(f);
    u += 0x7FFFu + ((u >> 16) & 1u);
    return (unsigned short)(u >> 16);
}

// pack two fp32 -> bf16x2 dword
__device__ __forceinline__ unsigned f2bf2(float a, float b) {
    return (unsigned)f2bf(a) | ((unsigned)f2bf(b) << 16);
}

// Build a v16bf WMMA fragment from LDS: two contiguous 8-element (16B) runs at
// +0 and +16 elements, per the CDNA5 16-bit A/B per-lane K layout.
__device__ __forceinline__ v16bf mkfrag(const unsigned short* p) {
    v8bf lo = *(const v8bf*)(p);
    v8bf hi = *(const v8bf*)(p + 16);
    v16bf r;
#pragma unroll
    for (int i = 0; i < 8; ++i) { r[i] = lo[i]; r[i + 8] = hi[i]; }
    return r;
}

// ---------------------------------------------------------------- init
__global__ void k_init(float* __restrict__ out, int* __restrict__ row_token,
                       int* __restrict__ counts) {
    int idx = blockIdx.x * blockDim.x + threadIdx.x;
    if (idx < TOKENS * HID) out[idx] = 0.f;
    if (idx < ROWS_PAD)     row_token[idx] = -1;
    if (idx < NEXP)         counts[idx] = 0;
}

// ---------------------------------------------------------------- router top-8 + softmax
__global__ void k_router(const float* __restrict__ logits, int* __restrict__ counts,
                         int* __restrict__ sel, float* __restrict__ wgt) {
    const int lane = threadIdx.x & 31;
    const int t = blockIdx.x * (NTHREADS / 32) + (threadIdx.x >> 5);
    if (t >= TOKENS) return;

    float v0 = logits[t * NEXP + lane];
    float v1 = logits[t * NEXP + 32 + lane];
    bool m0 = false, m1 = false;
    float tv[TOPK]; int ti[TOPK];

#pragma unroll
    for (int it = 0; it < TOPK; ++it) {
        float cand = -3.4e38f; int ci = 1 << 30;
        if (!m0) { cand = v0; ci = lane; }
        if (!m1 && (m0 || v1 > cand)) { cand = v1; ci = lane + 32; }
#pragma unroll
        for (int off = 16; off > 0; off >>= 1) {        // wave32 butterfly argmax
            float ov = __shfl_xor(cand, off);
            int   oi = __shfl_xor(ci, off);
            if (ov > cand || (ov == cand && oi < ci)) { cand = ov; ci = oi; }
        }
        tv[it] = cand; ti[it] = ci;
        if (ci == lane)      m0 = true;
        if (ci == lane + 32) m1 = true;
    }
    // softmax over the 8 selected (descending) logits
    float mx = tv[0], s = 0.f;
#pragma unroll
    for (int i = 0; i < TOPK; ++i) { tv[i] = __expf(tv[i] - mx); s += tv[i]; }
    float inv = 1.f / s;
    if (lane == 0) {
#pragma unroll
        for (int i = 0; i < TOPK; ++i) {
            sel[t * TOPK + i] = ti[i];
            wgt[t * TOPK + i] = tv[i] * inv;
            atomicAdd(&counts[ti[i]], 1);
        }
    }
}

// ---------------------------------------------------------------- tile-padded scan
__global__ void k_scan(const int* __restrict__ counts, int* __restrict__ off,
                       int* __restrict__ cursor) {
    if (threadIdx.x == 0 && blockIdx.x == 0) {
        int acc = 0;
        for (int e = 0; e < NEXP; ++e) {
            off[e] = acc; cursor[e] = acc;
            acc += (counts[e] + 31) & ~31;             // pad each group to 32 rows
        }
        off[NEXP] = acc;
    }
}

// ---------------------------------------------------------------- scatter into grouped order
__global__ void k_scatter(const int* __restrict__ sel, const float* __restrict__ wgt,
                          int* __restrict__ cursor, int* __restrict__ row_token,
                          float* __restrict__ row_wgt) {
    int idx = blockIdx.x * blockDim.x + threadIdx.x;
    if (idx >= ROWS) return;
    int t = idx / TOPK;
    int e = sel[idx];
    int pos = atomicAdd(&cursor[e], 1);
    row_token[pos] = t;
    row_wgt[pos]   = wgt[idx];
}

// ---------------------------------------------------------------- GEMM1: gate+up fused, SwiGLU
__global__ __launch_bounds__(NTHREADS)
void k_gemm1(const float* __restrict__ x, const float* __restrict__ Wgate,
             const float* __restrict__ Wup, const int* __restrict__ counts,
             const int* __restrict__ off, const int* __restrict__ row_token,
             unsigned short* __restrict__ h) {
    __shared__ unsigned short la[32][40];     // A tile (bf16), padded stride
    __shared__ unsigned short lbg[256][40];   // gate W tile, transposed [n][k]
    __shared__ unsigned short lbu[256][40];   // up   W tile, transposed [n][k]
    __shared__ int ltok[32];

    const int bt = blockIdx.x;
    const int total_tiles = off[NEXP] >> 5;
    if (bt >= total_tiles) return;
    const int rowbase = bt << 5;

    int e = 0;
#pragma unroll 1
    for (int i = 1; i < NEXP; ++i) if (off[i] <= rowbase) e = i;
    const int vend = off[e] + counts[e];
    const float* Wg = Wgate + (size_t)e * HID * INTER;
    const float* Wu = Wup   + (size_t)e * HID * INTER;
    const int n0 = blockIdx.y * 256;
    const int tid = threadIdx.x;

    if (tid < 32) ltok[tid] = row_token[rowbase + tid];
    __syncthreads();

    const int lane = tid & 31;
    const int w    = tid >> 5;
    const int wr   = w & 1;        // wave row (2 x 16 rows)
    const int wc   = w >> 1;       // wave col (4 x 64 cols)
    const int l16  = lane & 15;
    const int hi   = lane >> 4;

    const v8f zf = {0.f, 0.f, 0.f, 0.f, 0.f, 0.f, 0.f, 0.f};
    v8f accg[4] = {zf, zf, zf, zf};
    v8f accu[4] = {zf, zf, zf, zf};

    const int am = tid >> 3;               // A-load row 0..31
    const int aq = (tid & 7) << 2;         // A-load k quad
    const int atok = ltok[am];
    const float* xrow = x + (size_t)(atok >= 0 ? atok : 0) * HID;

    for (int k0 = 0; k0 < HID; k0 += 32) {
        // A: gather token row, fp32 -> bf16 (two packed dword stores)
        float4 av = make_float4(0.f, 0.f, 0.f, 0.f);
        if (atok >= 0) av = *(const float4*)(xrow + k0 + aq);
        *(unsigned*)&la[am][aq + 0] = f2bf2(av.x, av.y);
        *(unsigned*)&la[am][aq + 2] = f2bf2(av.z, av.w);
        // B: 32(k) x 256(n), coalesced over n, transposed store [n][k], packed x2
#pragma unroll 4
        for (int kk = 0; kk < 32; kk += 2) {
            const size_t go0 = (size_t)(k0 + kk) * INTER + n0 + tid;
            *(unsigned*)&lbg[tid][kk] = f2bf2(Wg[go0], Wg[go0 + INTER]);
            *(unsigned*)&lbu[tid][kk] = f2bf2(Wu[go0], Wu[go0 + INTER]);
        }
        if (k0 + 32 < HID) {
            __builtin_prefetch(Wg + (size_t)(k0 + 32) * INTER + n0 + tid, 0, 1);
            __builtin_prefetch(Wu + (size_t)(k0 + 32) * INTER + n0 + tid, 0, 1);
        }
        __syncthreads();

        const v16bf af = mkfrag(&la[wr * 16 + l16][hi * 8]);
#pragma unroll
        for (int j = 0; j < 4; ++j) {
            const int nn = wc * 64 + j * 16 + l16;
            const v16bf bg = mkfrag(&lbg[nn][hi * 8]);
            const v16bf bu = mkfrag(&lbu[nn][hi * 8]);
            accg[j] = __builtin_amdgcn_wmma_f32_16x16x32_bf16(
                false, af, false, bg, (short)0, accg[j], false, false);
            accu[j] = __builtin_amdgcn_wmma_f32_16x16x32_bf16(
                false, af, false, bu, (short)0, accu[j], false, false);
        }
        __syncthreads();
    }

    // epilogue: h = silu(gate) * up  (bf16)
    const int mloc = wr * 16 + hi * 8;
#pragma unroll
    for (int j = 0; j < 4; ++j) {
        const int nn = n0 + wc * 64 + j * 16 + l16;
#pragma unroll
        for (int v = 0; v < 8; ++v) {
            const int r = rowbase + mloc + v;
            if (r < vend) {
                float g = accg[j][v], u = accu[j][v];
                float s = g / (1.f + __expf(-g));
                h[(size_t)r * INTER + nn] = f2bf(s * u);
            }
        }
    }
}

// ---------------------------------------------------------------- GEMM2: down proj + combine
__global__ __launch_bounds__(NTHREADS)
void k_gemm2(const unsigned short* __restrict__ h, const float* __restrict__ Wdown,
             const int* __restrict__ counts, const int* __restrict__ off,
             const int* __restrict__ row_token, const float* __restrict__ row_wgt,
             float* __restrict__ out) {
    __shared__ unsigned short la[32][40];
    __shared__ unsigned short lb[256][40];
    __shared__ int   ltok[32];
    __shared__ float lw[32];

    const int bt = blockIdx.x;
    const int total_tiles = off[NEXP] >> 5;
    if (bt >= total_tiles) return;
    const int rowbase = bt << 5;

    int e = 0;
#pragma unroll 1
    for (int i = 1; i < NEXP; ++i) if (off[i] <= rowbase) e = i;
    const int vend = off[e] + counts[e];
    const float* Wd = Wdown + (size_t)e * INTER * HID;
    const int n0 = blockIdx.y * 256;
    const int tid = threadIdx.x;

    if (tid < 32) { ltok[tid] = row_token[rowbase + tid]; lw[tid] = row_wgt[rowbase + tid]; }
    __syncthreads();

    const int lane = tid & 31;
    const int w    = tid >> 5;
    const int wr   = w & 1;
    const int wc   = w >> 1;
    const int l16  = lane & 15;
    const int hi   = lane >> 4;

    const v8f zf = {0.f, 0.f, 0.f, 0.f, 0.f, 0.f, 0.f, 0.f};
    v8f acc[4] = {zf, zf, zf, zf};

    const int am = tid >> 3;
    const int aq = (tid & 7) << 2;
    const unsigned short* hrow = h + (size_t)(rowbase + am) * INTER;
    // LDS byte address for this thread's 8-byte A-tile slot (generic-pointer
    // low 32 bits == LDS offset on AMDGPU)
    const unsigned lds_a = (unsigned)(uintptr_t)&la[am][aq];

    for (int k0 = 0; k0 < INTER; k0 += 32) {
        // A: h rows are already bf16 -> async copy global->LDS, 8B per lane,
        // tracked by ASYNCcnt (no VGPR round-trip, no conversion needed)
        asm volatile("global_load_async_to_lds_b64 %0, %1, off"
                     :: "v"(lds_a), "v"(hrow + k0 + aq) : "memory");
        // B: fp32 weights, convert + transpose into LDS, packed x2
#pragma unroll 4
        for (int kk = 0; kk < 32; kk += 2) {
            const size_t go0 = (size_t)(k0 + kk) * HID + n0 + tid;
            *(unsigned*)&lb[tid][kk] = f2bf2(Wd[go0], Wd[go0 + HID]);
        }
        if (k0 + 32 < INTER)
            __builtin_prefetch(Wd + (size_t)(k0 + 32) * HID + n0 + tid, 0, 1);
        asm volatile("s_wait_asynccnt 0x0" ::: "memory");
        __syncthreads();

        const v16bf af = mkfrag(&la[wr * 16 + l16][hi * 8]);
#pragma unroll
        for (int j = 0; j < 4; ++j) {
            const v16bf bf = mkfrag(&lb[wc * 64 + j * 16 + l16][hi * 8]);
            acc[j] = __builtin_amdgcn_wmma_f32_16x16x32_bf16(
                false, af, false, bf, (short)0, acc[j], false, false);
        }
        __syncthreads();
    }

    // epilogue: scale by routing weight, combine into out[token]
    const int mloc = wr * 16 + hi * 8;
#pragma unroll
    for (int j = 0; j < 4; ++j) {
        const int nn = n0 + wc * 64 + j * 16 + l16;
#pragma unroll
        for (int v = 0; v < 8; ++v) {
            const int r = rowbase + mloc + v;
            if (r < vend) {
                const int tok = ltok[mloc + v];
                atomicAdd(&out[(size_t)tok * HID + nn], acc[j][v] * lw[mloc + v]);
            }
        }
    }
}

// ---------------------------------------------------------------- launch
extern "C" void kernel_launch(void* const* d_in, const int* in_sizes, int n_in,
                              void* d_out, int out_size, void* d_ws, size_t ws_size,
                              hipStream_t stream) {
    (void)in_sizes; (void)n_in; (void)out_size; (void)ws_size;
    const float* x      = (const float*)d_in[0];   // [T, H]
    const float* logits = (const float*)d_in[1];   // [T, E]
    const float* Wg     = (const float*)d_in[2];   // [E, H, I]
    const float* Wu     = (const float*)d_in[3];   // [E, H, I]
    const float* Wd     = (const float*)d_in[4];   // [E, I, H]
    float* out = (float*)d_out;                    // [T, H]

    char* ws = (char*)d_ws;
    int*   counts    = (int*)ws;            ws += 64 * 4;
    int*   off       = (int*)ws;            ws += 72 * 4;        // 65 used, padded
    int*   cursor    = (int*)ws;            ws += 64 * 4;
    int*   sel       = (int*)ws;            ws += (size_t)ROWS * 4;
    float* wgt       = (float*)ws;          ws += (size_t)ROWS * 4;
    int*   row_token = (int*)ws;            ws += (size_t)ROWS_PAD * 4;
    float* row_wgt   = (float*)ws;          ws += (size_t)ROWS_PAD * 4;
    unsigned short* h = (unsigned short*)ws;  // [ROWS_PAD, INTER] bf16

    k_init<<<(TOKENS * HID + NTHREADS - 1) / NTHREADS, NTHREADS, 0, stream>>>(
        out, row_token, counts);
    k_router<<<TOKENS / (NTHREADS / 32), NTHREADS, 0, stream>>>(logits, counts, sel, wgt);
    k_scan<<<1, 1, 0, stream>>>(counts, off, cursor);
    k_scatter<<<ROWS / NTHREADS, NTHREADS, 0, stream>>>(sel, wgt, cursor, row_token, row_wgt);

    dim3 g1(MAX_TILES, INTER / 256);
    k_gemm1<<<g1, NTHREADS, 0, stream>>>(x, Wg, Wu, counts, off, row_token, h);
    dim3 g2(MAX_TILES, HID / 256);
    k_gemm2<<<g2, NTHREADS, 0, stream>>>(h, Wd, counts, off, row_token, row_wgt, out);
}
